// MultiSpectralFFTLayer_87067577025395
// MI455X (gfx1250) — matrix operand order repaired and checked by reading, not compiled, not run on hardware
//
#include <hip/hip_runtime.h>
#include <hip/hip_bf16.h>
#include <stdint.h>

#define CH    256
#define FH    56
#define FW    56
#define PLANE (FH*FW)      // 3136 floats per (n,c) plane
#define NB    64
#define RED   16
#define HID   (CH/RED)     // 16

typedef __attribute__((ext_vector_type(16))) __bf16 v16bf;
typedef __attribute__((ext_vector_type(8)))  float  v8f;

// A-operand (16-bit, 16x32) K index for element e, lane-half hf:
//   VGPR0..3: K = e + 8*hf  (e<8) ; VGPR4..7: K = e + 8 + 8*hf (e>=8)
__device__ __forceinline__ int aK(int e, int hf) { return (e < 8 ? e : e + 8) + 8 * hf; }

// ---------------------------------------------------------------------------
// K1: spectral pooling.  Exact identity:
//   sum_{h,w} Re ifft2(ifftshift(fftshift(fft2 x)*mask)) = mask[c,28,28]*sum_{h,w} x
// Stream the plane into LDS with async-to-LDS b128 loads, then tree-reduce.
// ---------------------------------------------------------------------------
__global__ void k_pool(const float* __restrict__ x, const float* __restrict__ mask,
                       float* __restrict__ y) {
  __shared__ float tile[PLANE];
  __shared__ float part[8];
  const int nc = blockIdx.x;
  const int t  = threadIdx.x;
  const float* xp = x + (size_t)nc * PLANE;

  const uint32_t lds0 = (uint32_t)(uintptr_t)tile;   // low 32 bits of flat LDS addr = LDS offset
  for (int i = t; i < PLANE / 4; i += 256) {
    uint32_t la = lds0 + (uint32_t)i * 16u;
    uint64_t ga = (uint64_t)(uintptr_t)xp + (uint64_t)i * 16u;
    asm volatile("global_load_async_to_lds_b128 %0, %1, off"
                 :: "v"(la), "v"(ga) : "memory");
  }
  asm volatile("s_wait_asynccnt 0x0" ::: "memory");
  __syncthreads();

  float s = 0.f;
  for (int i = t; i < PLANE; i += 256) s += tile[i];
#pragma unroll
  for (int off = 16; off > 0; off >>= 1) s += __shfl_down(s, off, 32);
  if ((t & 31) == 0) part[t >> 5] = s;
  __syncthreads();
  if (t == 0) {
    float tot = 0.f;
#pragma unroll
    for (int w = 0; w < 8; ++w) tot += part[w];
    const int c = nc & (CH - 1);
    y[nc] = tot * mask[(size_t)c * PLANE + 28 * FW + 28];
  }
}

// ---------------------------------------------------------------------------
// K2: SE gate  g = sigmoid( relu(y @ w1^T) @ w2^T ),  y:[64,256] w1:[16,256] w2:[256,16]
// One block, 4 waves; wave w owns rows [16w,16w+16). All matrix math on WMMA.
// ---------------------------------------------------------------------------
__global__ void k_gate(const float* __restrict__ y, const float* __restrict__ w1,
                       const float* __restrict__ w2, float* __restrict__ g) {
  __shared__ float hsm[NB][HID];
  const int lane = threadIdx.x & 31;
  const int wave = threadIdx.x >> 5;
  const int hf   = lane >> 4;      // lane half (0: lanes 0-15, 1: lanes 16-31)
  const int mn   = lane & 15;      // row (A) / col (B,D) index within tile
  const int rowBase = wave * 16;

  // ---- GEMM1: h = relu(y @ w1^T), K = 256 in 8 steps of 32 ----
  v8f acc = {};
#pragma unroll
  for (int kb = 0; kb < CH / 32; ++kb) {
    v16bf a, b;
#pragma unroll
    for (int e = 0; e < 16; ++e) {
      a[e] = (__bf16)y[(rowBase + mn) * CH + kb * 32 + aK(e, hf)];
      // B[k][n] = w1[n][k]; B layout: lane col n=mn, K = aK(e, hf)
      b[e] = (__bf16)w1[mn * CH + kb * 32 + aK(e, hf)];
    }
    acc = __builtin_amdgcn_wmma_f32_16x16x32_bf16(false, a, false, b,
                                                  (short)0, acc, false, false);
  }
  // D layout: VGPR r, lane -> (m = r + 8*hf, n = mn). ReLU, restage via LDS.
#pragma unroll
  for (int r = 0; r < 8; ++r) {
    float v = acc[r];
    hsm[rowBase + r + 8 * hf][mn] = v > 0.f ? v : 0.f;
  }
  __syncthreads();

  // ---- GEMM2: g = sigmoid(h @ w2^T), K = 16 zero-padded to 32 ----
  v16bf a2;
#pragma unroll
  for (int e = 0; e < 16; ++e) {
    int k = aK(e, hf);
    a2[e] = (k < HID) ? (__bf16)hsm[rowBase + mn][k] : (__bf16)0.f;
  }
#pragma unroll
  for (int j = 0; j < CH / 16; ++j) {
    v16bf b2;
#pragma unroll
    for (int e = 0; e < 16; ++e) {
      int k = aK(e, hf);
      // B[k][n] = w2[n][k], n = 16j + mn
      b2[e] = (k < HID) ? (__bf16)w2[(j * 16 + mn) * HID + k] : (__bf16)0.f;
    }
    v8f c2 = {};
    c2 = __builtin_amdgcn_wmma_f32_16x16x32_bf16(false, a2, false, b2,
                                                 (short)0, c2, false, false);
#pragma unroll
    for (int r = 0; r < 8; ++r) {
      float v = c2[r];
      g[(rowBase + r + 8 * hf) * CH + j * 16 + mn] = 1.f / (1.f + __expf(-v));
    }
  }
}

// ---------------------------------------------------------------------------
// K3: out = x * g[n,c]  (float4 streaming; second read of x mostly L2-resident)
// ---------------------------------------------------------------------------
__global__ void k_scale(const float* __restrict__ x, const float* __restrict__ g,
                        float* __restrict__ out) {
  const int nc = blockIdx.x;
  const float gv = g[nc];
  const float4* xp = (const float4*)(x + (size_t)nc * PLANE);
  float4* op = (float4*)(out + (size_t)nc * PLANE);
  for (int i = threadIdx.x; i < PLANE / 4; i += 256) {
    float4 v = xp[i];
    v.x *= gv; v.y *= gv; v.z *= gv; v.w *= gv;
    op[i] = v;
  }
}

extern "C" void kernel_launch(void* const* d_in, const int* in_sizes, int n_in,
                              void* d_out, int out_size, void* d_ws, size_t ws_size,
                              hipStream_t stream) {
  const float* x    = (const float*)d_in[0];   // [64,256,56,56]
  const float* w1   = (const float*)d_in[1];   // [16,256]
  const float* w2   = (const float*)d_in[2];   // [256,16]
  const float* mask = (const float*)d_in[3];   // [256,56,56]
  float* y = (float*)d_ws;                     // [64*256]
  float* g = y + NB * CH;                      // [64*256]
  float* out = (float*)d_out;

  k_pool <<<NB * CH, 256, 0, stream>>>(x, mask, y);
  k_gate <<<1, 128, 0, stream>>>(y, w1, w2, g);
  k_scale<<<NB * CH, 256, 0, stream>>>(x, g, out);
}
